// StructureGNN_15487652069425
// MI455X (gfx1250) — matrix operand order repaired and checked by reading, not compiled, not run on hardware
//
#include <hip/hip_runtime.h>
#include <hip/hip_bf16.h>

typedef __attribute__((ext_vector_type(16))) _Float16 v16h;
typedef __attribute__((ext_vector_type(8)))  float    v8f;
typedef __attribute__((ext_vector_type(4)))  int      v4i;

#define FIN 20
#define HID 128
#define OUTC 64

// gfx1250 async global->LDS (ASYNCcnt-tracked, bypasses VGPRs). Guarded so the
// compile stays green on toolchains without the builtins.
#if defined(__has_builtin)
#  if __has_builtin(__builtin_amdgcn_global_load_async_to_lds_b128) && \
      __has_builtin(__builtin_amdgcn_s_wait_asynccnt)
#    define HAVE_ASYNC_LDS 1
#  endif
#endif
#ifndef HAVE_ASYNC_LDS
#  define HAVE_ASYNC_LDS 0
#endif

typedef __attribute__((address_space(1))) v4i gv4i;   // global 128-bit chunk
typedef __attribute__((address_space(3))) v4i lv4i;   // LDS 128-bit chunk

// ---------------------------------------------------------------------------
// WMMA GEMM: Out[nrows, NCOLS] = A[nrows, Kvalid] @ W[Kvalid, NCOLS] (+bias)
// f32 in -> f16 fragments -> v_wmma_f32_16x16x32_f16 (f32 accumulate) -> f32 out
// One block = 256 threads = 8 waves; block computes a 128-row tile.
// A-slices staged to LDS with GLOBAL_LOAD_ASYNC_TO_LDS_B128 when fully in-bounds.
// ---------------------------------------------------------------------------
template<int KP, int NCOLS, bool HAS_BIAS>
__global__ __launch_bounds__(256)
void gemm_wmma(const float* __restrict__ A, int lda, int Kvalid,
               const float* __restrict__ W,          // [Kvalid, NCOLS] row-major
               const float* __restrict__ bias,
               float* __restrict__ Out, int nrows)
{
    __shared__ _Float16 sW[NCOLS * KP];   // transposed: sW[n*KP + k]
    __shared__ float    sA[128 * 32];     // current 32-wide K slice, f32

    const int tid  = threadIdx.x;
    const int wave = tid >> 5;
    const int lane = tid & 31;
    const int half = lane >> 4;
    const int r    = lane & 15;
    const int row0 = blockIdx.x * 128;

    // Stage all weights into LDS (f16, transposed for contiguous per-column K)
    for (int i = tid; i < KP * NCOLS; i += 256) {
        int k = i / NCOLS, n = i % NCOLS;            // coalesced global read
        float v = (k < Kvalid) ? W[(size_t)k * NCOLS + n] : 0.0f;
        sW[n * KP + k] = (_Float16)v;
    }

    constexpr int NT = NCOLS / 16;
    v8f acc[NT];
    #pragma unroll
    for (int nt = 0; nt < NT; ++nt) acc[nt] = v8f{};

    const bool tileFull = (row0 + 128 <= nrows) && (Kvalid == KP);

    for (int k0 = 0; k0 < KP; k0 += 32) {
        __syncthreads();
#if HAVE_ASYNC_LDS
        if (tileFull) {
            // 128 rows x 32 floats = 1024 x b128 chunks, 4 per thread, no VGPR staging
            for (int i = tid; i < 128 * 8; i += 256) {
                int rr = i >> 3, cc = (i & 7) << 2;
                const float* gp = A + (size_t)(row0 + rr) * lda + (k0 + cc);
                __builtin_amdgcn_global_load_async_to_lds_b128(
                    (gv4i*)(void*)gp, (lv4i*)(void*)&sA[rr * 32 + cc], 0, 0);
            }
            __builtin_amdgcn_s_wait_asynccnt(0);
        } else
#endif
        {
            for (int i = tid; i < 128 * 32; i += 256) {
                int rr = i >> 5, cc = i & 31;
                int grow = row0 + rr, gk = k0 + cc;
                sA[rr * 32 + cc] = (grow < nrows && gk < Kvalid)
                                 ? A[(size_t)grow * lda + gk] : 0.0f;
            }
        }
        __syncthreads();

        // A fragment (ISA 16-bit A 16x32 layout): row M = wave*16 + r
        v16h af;
        const int arow = wave * 16 + r;
        #pragma unroll
        for (int vv = 0; vv < 8; ++vv) {
            int kb = 2 * (vv & 3) + 16 * (vv >> 2) + 8 * half;   // K base for this VGPR
            af[2 * vv]     = (_Float16)sA[arow * 32 + kb];
            af[2 * vv + 1] = (_Float16)sA[arow * 32 + kb + 1];
        }

        #pragma unroll
        for (int nt = 0; nt < NT; ++nt) {
            // B fragment (32x16): lane column = nt*16 + r, K = k0 + (e&7) + 8*half + 16*(e>>3)
            v16h bf;
            const int col = nt * 16 + r;
            #pragma unroll
            for (int e = 0; e < 16; ++e)
                bf[e] = sW[col * KP + k0 + (e & 7) + 8 * half + 16 * (e >> 3)];

            acc[nt] = __builtin_amdgcn_wmma_f32_16x16x32_f16(
                false, af, false, bf, (short)0, acc[nt], false, false);
        }
    }

    // Preload bias once per lane-column (no loads/branches in the store loop)
    float bv[NT];
    #pragma unroll
    for (int nt = 0; nt < NT; ++nt) bv[nt] = HAS_BIAS ? bias[nt * 16 + r] : 0.0f;

    // C/D layout: VGPR j -> M = j + 8*half, N = r
    const int mbase = row0 + wave * 16 + 8 * half;
    if (mbase + 8 <= nrows) {                        // fast path: full tile
        #pragma unroll
        for (int nt = 0; nt < NT; ++nt) {
            #pragma unroll
            for (int j = 0; j < 8; ++j) {
                float v = acc[nt][j];
                if (HAS_BIAS) v += bv[nt];           // compiled out when !HAS_BIAS
                Out[(size_t)(mbase + j) * NCOLS + nt * 16 + r] = v;
            }
        }
    } else {
        #pragma unroll
        for (int nt = 0; nt < NT; ++nt) {
            #pragma unroll
            for (int j = 0; j < 8; ++j) {
                if (mbase + j < nrows) {
                    float v = acc[nt][j];
                    if (HAS_BIAS) v += bv[nt];
                    Out[(size_t)(mbase + j) * NCOLS + nt * 16 + r] = v;
                }
            }
        }
    }
}

// ---------------------------------------------------------------------------
// Degree / normalization
// ---------------------------------------------------------------------------
__global__ void deg_init(float* __restrict__ deg, int n) {
    int i = blockIdx.x * blockDim.x + threadIdx.x;
    if (i < n) deg[i] = 1.0f;                         // self-loop contribution
}
__global__ void deg_edges(const int* __restrict__ dst, int ne, float* __restrict__ deg) {
    int e = blockIdx.x * blockDim.x + threadIdx.x;
    if (e < ne) atomicAdd(&deg[dst[e]], 1.0f);
}
__global__ void deg_rsqrt(float* __restrict__ deg, int n) {
    int i = blockIdx.x * blockDim.x + threadIdx.x;
    if (i < n) deg[i] = rsqrtf(deg[i]);               // deg >= 1 always
}

// ---------------------------------------------------------------------------
// hout[i,:] = dinv[i]^2 * xw[i,:]   (self-loop term; also zero-initializes hout)
// ---------------------------------------------------------------------------
__global__ void selfloop_init(const float* __restrict__ xw, const float* __restrict__ dinv,
                              float* __restrict__ hout, int n) {
    size_t idx = (size_t)blockIdx.x * blockDim.x + threadIdx.x;
    size_t total = (size_t)n * HID;
    if (idx < total) {
        int row = (int)(idx >> 7);
        float d = dinv[row];
        hout[idx] = xw[idx] * d * d;
    }
}

// One wave per edge: 32 lanes x float4 = 128 floats, coalesced gather + f32 atomics
__global__ void edge_scatter(const int* __restrict__ src, const int* __restrict__ dst, int ne,
                             const float* __restrict__ xw, const float* __restrict__ dinv,
                             float* __restrict__ hout) {
    int e = blockIdx.x * (blockDim.x >> 5) + (threadIdx.x >> 5);
    if (e >= ne) return;
    int lane = threadIdx.x & 31;
    int s = src[e], d = dst[e];
    float coef = dinv[s] * dinv[d];
    const float4* xs = (const float4*)(xw + (size_t)s * HID);
    float4 v = xs[lane];
    float* out = hout + (size_t)d * HID + lane * 4;
    atomicAdd(out + 0, v.x * coef);
    atomicAdd(out + 1, v.y * coef);
    atomicAdd(out + 2, v.z * coef);
    atomicAdd(out + 3, v.w * coef);
}

__global__ void bias_relu(float* __restrict__ h, const float* __restrict__ b, size_t total) {
    size_t idx = (size_t)blockIdx.x * blockDim.x + threadIdx.x;
    if (idx < total) {
        int c = (int)(idx & (HID - 1));
        float v = h[idx] + b[c];
        h[idx] = v > 0.0f ? v : 0.0f;
    }
}

// ---------------------------------------------------------------------------
// Mean pool over graph ids
// ---------------------------------------------------------------------------
__global__ void pool_zero(float* __restrict__ pooled, float* __restrict__ counts, int g) {
    int i = blockIdx.x * blockDim.x + threadIdx.x;
    if (i < g * HID) pooled[i] = 0.0f;
    if (i < g) counts[i] = 0.0f;
}
__global__ void pool_accum(const float* __restrict__ h, const int* __restrict__ batch, int n,
                           float* __restrict__ pooled, float* __restrict__ counts) {
    int row = blockIdx.x * (blockDim.x >> 5) + (threadIdx.x >> 5);
    if (row >= n) return;
    int lane = threadIdx.x & 31;
    int g = batch[row];
    const float4* hv = (const float4*)(h + (size_t)row * HID);
    float4 v = hv[lane];
    float* p = pooled + (size_t)g * HID + lane * 4;
    atomicAdd(p + 0, v.x);
    atomicAdd(p + 1, v.y);
    atomicAdd(p + 2, v.z);
    atomicAdd(p + 3, v.w);
    if (lane == 0) atomicAdd(&counts[g], 1.0f);
}
__global__ void pool_div(float* __restrict__ pooled, const float* __restrict__ counts, int g) {
    int idx = blockIdx.x * blockDim.x + threadIdx.x;
    if (idx < g * HID) {
        int gg = idx >> 7;
        pooled[idx] /= fmaxf(counts[gg], 1.0f);
    }
}

// ---------------------------------------------------------------------------
static inline size_t alignUp(size_t x) { return (x + 255) & ~(size_t)255; }

extern "C" void kernel_launch(void* const* d_in, const int* in_sizes, int n_in,
                              void* d_out, int out_size, void* d_ws, size_t ws_size,
                              hipStream_t stream) {
    const float* x    = (const float*)d_in[0];
    const int*   ei   = (const int*)d_in[1];
    const int*   batch= (const int*)d_in[2];
    const float* W1   = (const float*)d_in[3];
    const float* b1   = (const float*)d_in[4];
    const float* W2   = (const float*)d_in[5];
    const float* b2   = (const float*)d_in[6];
    const float* W3   = (const float*)d_in[7];
    const float* b3   = (const float*)d_in[8];
    const float* Wout = (const float*)d_in[9];
    const float* bout = (const float*)d_in[10];
    float* out = (float*)d_out;

    const int N = in_sizes[0] / FIN;          // 100000
    const int E = in_sizes[1] / 2;            // 1600000
    const int G = out_size / OUTC;            // 512
    const int* src = ei;
    const int* dst = ei + E;

    // Workspace carve-up (floats)
    float* ws = (float*)d_ws;
    size_t off = 0;
    float* dinv   = ws + off; off = alignUp(off + (size_t)N);
    float* xw     = ws + off; off = alignUp(off + (size_t)N * HID);
    float* hA     = ws + off; off = alignUp(off + (size_t)N * HID);
    float* hB     = ws + off; off = alignUp(off + (size_t)N * HID);
    float* pooled = ws + off; off = alignUp(off + (size_t)G * HID);
    float* counts = ws + off; off = alignUp(off + (size_t)G);
    (void)ws_size; (void)n_in;

    const size_t NH = (size_t)N * HID;
    const int TPB = 256;
    const int gElem   = (int)((NH + TPB - 1) / TPB);
    const int gNode   = (N + TPB - 1) / TPB;
    const int gEdgeT  = (E + TPB - 1) / TPB;
    const int gEdgeW  = (E + 7) / 8;          // one wave (of 8/block) per edge
    const int gNodeW  = (N + 7) / 8;
    const int gGemm   = (N + 127) / 128;

    // --- symmetric normalization coefficients ---
    deg_init<<<gNode, TPB, 0, stream>>>(dinv, N);
    deg_edges<<<gEdgeT, TPB, 0, stream>>>(dst, E, dinv);
    deg_rsqrt<<<gNode, TPB, 0, stream>>>(dinv, N);

    // --- layer 1: K=20 padded to 32 ---
    gemm_wmma<32, HID, false><<<gGemm, TPB, 0, stream>>>(x, FIN, FIN, W1, nullptr, xw, N);
    selfloop_init<<<gElem, TPB, 0, stream>>>(xw, dinv, hA, N);
    edge_scatter<<<gEdgeW, TPB, 0, stream>>>(src, dst, E, xw, dinv, hA);
    bias_relu<<<gElem, TPB, 0, stream>>>(hA, b1, NH);

    // --- layer 2 ---
    gemm_wmma<HID, HID, false><<<gGemm, TPB, 0, stream>>>(hA, HID, HID, W2, nullptr, xw, N);
    selfloop_init<<<gElem, TPB, 0, stream>>>(xw, dinv, hB, N);
    edge_scatter<<<gEdgeW, TPB, 0, stream>>>(src, dst, E, xw, dinv, hB);
    bias_relu<<<gElem, TPB, 0, stream>>>(hB, b2, NH);

    // --- layer 3 ---
    gemm_wmma<HID, HID, false><<<gGemm, TPB, 0, stream>>>(hB, HID, HID, W3, nullptr, xw, N);
    selfloop_init<<<gElem, TPB, 0, stream>>>(xw, dinv, hA, N);
    edge_scatter<<<gEdgeW, TPB, 0, stream>>>(src, dst, E, xw, dinv, hA);
    bias_relu<<<gElem, TPB, 0, stream>>>(hA, b3, NH);

    // --- global mean pool ---
    pool_zero<<<(G * HID + TPB - 1) / TPB, TPB, 0, stream>>>(pooled, counts, G);
    pool_accum<<<gNodeW, TPB, 0, stream>>>(hA, batch, N, pooled, counts);
    pool_div<<<(G * HID + TPB - 1) / TPB, TPB, 0, stream>>>(pooled, counts, G);

    // --- output head: [G,128] @ [128,64] + bout ---
    gemm_wmma<HID, OUTC, true><<<(G + 127) / 128, TPB, 0, stream>>>(pooled, HID, HID, Wout, bout, out, G);
}
